// MixtralMoE_32607391711910
// MI455X (gfx1250) — matrix operand order repaired and checked by reading, not compiled, not run on hardware
//
#include <hip/hip_runtime.h>
#include <math.h>

#define HID 2048
#define FFN_ 5632
#define NTOK 2048
#define NEXP 8
#define MAXROWS 4096  // NTOK * TOP_K

typedef __attribute__((ext_vector_type(16))) __bf16 v16bf;
typedef __attribute__((ext_vector_type(16))) float  v16f;
typedef __attribute__((ext_vector_type(8)))  float  v8f;

union FragU { v16bf v; unsigned u[8]; uint4 q[2]; };

// Native bf16 conversion (lowers to v_cvt_pk_bf16_f32).
static __device__ __forceinline__ unsigned pk2(float a, float b) {
  union { __bf16 h[2]; unsigned u; } c;
  c.h[0] = (__bf16)a;
  c.h[1] = (__bf16)b;
  return c.u;
}

// B fragment (32x16 bf16): lane-half holds 16 consecutive K values starting at kb.
// Vector fptrunc -> 8x v_cvt_pk_bf16_f32, packing already in lane order (no perms).
static __device__ __forceinline__ FragU load_b_row(const float* __restrict__ r, int kb) {
  union { v16f f; float4 q[4]; } t;
  t.q[0] = *(const float4*)(r + kb);
  t.q[1] = *(const float4*)(r + kb + 4);
  t.q[2] = *(const float4*)(r + kb + 8);
  t.q[3] = *(const float4*)(r + kb + 12);
  FragU b;
  b.v = __builtin_convertvector(t.f, v16bf);
  return b;
}

#define WMMA_BF16(A, B, C) \
  __builtin_amdgcn_wmma_f32_16x16x32_bf16(false, (A), false, (B), (short)0, (C), false, false)

// ---------------------------------------------------------------------------
// K1: router — one wave per token. logits = x @ wg^T, top-2 + renorm weights.
// ---------------------------------------------------------------------------
__global__ __launch_bounds__(256) void moe_router(const float* __restrict__ x,
                                                  const float* __restrict__ wg,
                                                  int* __restrict__ sel,
                                                  float* __restrict__ selw) {
  const int lane = threadIdx.x & 31;
  const int wid  = threadIdx.x >> 5;
  const int t = blockIdx.x * 8 + wid;
  const float* xr = x + (size_t)t * HID;
  float acc[NEXP];
#pragma unroll
  for (int e = 0; e < NEXP; ++e) acc[e] = 0.f;
  for (int k = lane; k < HID; k += 32) {
    const float xv = xr[k];
#pragma unroll
    for (int e = 0; e < NEXP; ++e) acc[e] += xv * wg[e * HID + k];
  }
#pragma unroll
  for (int off = 16; off > 0; off >>= 1) {
#pragma unroll
    for (int e = 0; e < NEXP; ++e) acc[e] += __shfl_xor(acc[e], off, 32);
  }
  if (lane == 0) {
    int i0 = 0;
#pragma unroll
    for (int e = 1; e < NEXP; ++e) if (acc[e] > acc[i0]) i0 = e;
    int i1 = (i0 == 0) ? 1 : 0;
#pragma unroll
    for (int e = 0; e < NEXP; ++e) if (e != i0 && acc[e] > acc[i1]) i1 = e;
    const float p1 = __expf(acc[i1] - acc[i0]);
    const float inv = 1.f / (1.f + p1);
    sel[t * 2]     = i0;  selw[t * 2]     = inv;
    sel[t * 2 + 1] = i1;  selw[t * 2 + 1] = p1 * inv;
  }
}

// ---------------------------------------------------------------------------
// K2: build compacted per-expert token lists (single block).
// ---------------------------------------------------------------------------
__global__ __launch_bounds__(256) void moe_build(const int* __restrict__ sel,
                                                 const float* __restrict__ selw,
                                                 int* __restrict__ cnts_g,
                                                 int* __restrict__ offs_g,
                                                 int* __restrict__ tok,
                                                 float* __restrict__ gw) {
  __shared__ int cnt[NEXP], off[NEXP], cur[NEXP];
  const int tid = threadIdx.x;
  if (tid < NEXP) { cnt[tid] = 0; cur[tid] = 0; }
  __syncthreads();
  for (int i = tid; i < MAXROWS; i += 256) atomicAdd(&cnt[sel[i]], 1);
  __syncthreads();
  if (tid == 0) {
    int s = 0;
    for (int e = 0; e < NEXP; ++e) { off[e] = s; s += cnt[e]; }
  }
  __syncthreads();
  for (int i = tid; i < MAXROWS; i += 256) {
    const int e = sel[i];
    const int slot = atomicAdd(&cur[e], 1);
    const int g = off[e] + slot;
    tok[g] = i >> 1;
    gw[g]  = selw[i];
  }
  if (tid < NEXP) { cnts_g[tid] = cnt[tid]; offs_g[tid] = off[tid]; }
}

// ---------------------------------------------------------------------------
// K3: zero output (poisoned by harness).
// ---------------------------------------------------------------------------
__global__ __launch_bounds__(256) void moe_zero(float* __restrict__ o) {
  const size_t i = ((size_t)blockIdx.x * 256 + threadIdx.x) * 4;
  *(float4*)(o + i) = float4{0.f, 0.f, 0.f, 0.f};
}

// ---------------------------------------------------------------------------
// K4: pre-convert activations to bf16 once (8 floats / thread).
// ---------------------------------------------------------------------------
__global__ __launch_bounds__(256) void moe_cvt_x(const float* __restrict__ x,
                                                 unsigned* __restrict__ xb) {
  const size_t i = ((size_t)blockIdx.x * 256 + threadIdx.x) * 8;
  const float4 f0 = *(const float4*)(x + i);
  const float4 f1 = *(const float4*)(x + i + 4);
  uint4 q;
  q.x = pk2(f0.x, f0.y); q.y = pk2(f0.z, f0.w);
  q.z = pk2(f1.x, f1.y); q.w = pk2(f1.z, f1.w);
  *(uint4*)(xb + i / 2) = q;
}

// ---------------------------------------------------------------------------
// K5: up projection.  h[row] = silu(x@w1e^T) * (x@w3e^T)   (bf16 out)
// block tile: 128 tokens x 64 FFN, 8 waves (4 M x 2 N),
// each wave: 2 M subtiles x 2 N subtiles x 2 matrices = 8 WMMAs / K-step,
// every B fragment is shared by two WMMAs (M reuse).
// ---------------------------------------------------------------------------
__global__ __launch_bounds__(256) void moe_up(const unsigned short* __restrict__ xbf,
                                              const float* __restrict__ w1,
                                              const float* __restrict__ w3,
                                              const int* __restrict__ cnts,
                                              const int* __restrict__ offs,
                                              const int* __restrict__ tok,
                                              __bf16* __restrict__ h) {
  const int e   = blockIdx.z;
  const int cnt = cnts[e];
  const int t0  = blockIdx.y * 128;
  if (t0 >= cnt) return;
  const int goff = offs[e] + t0;
  const int nb   = blockIdx.x * 64;

  const int tid  = threadIdx.x;
  const int lane = tid & 31, wid = tid >> 5;
  const int l15 = lane & 15, half = lane >> 4;
  const int waveM = wid >> 1, waveN = wid & 1;

  __shared__ unsigned sxu[128][128];  // 128 rows x 256 bf16 of x-chunk (64KB)

  const size_t wbase = (size_t)e * FFN_ * HID;
  const int n0 = nb + waveN * 32 + l15;
  const int n1 = n0 + 16;
  const float* w1r0 = w1 + wbase + (size_t)n0 * HID;
  const float* w1r1 = w1 + wbase + (size_t)n1 * HID;
  const float* w3r0 = w3 + wbase + (size_t)n0 * HID;
  const float* w3r1 = w3 + wbase + (size_t)n1 * HID;

  v8f c1m0n0 = {}, c1m1n0 = {}, c1m0n1 = {}, c1m1n1 = {};
  v8f c3m0n0 = {}, c3m1n0 = {}, c3m0n1 = {}, c3m1n1 = {};

  // staging: row = tid/2, 128 consecutive bf16 cols per thread
  const int sr = tid >> 1;
  const int sc = (tid & 1) * 128;
  int gRow = goff + sr;
  if (t0 + sr >= cnt) gRow = goff;  // clamp padding rows to a valid row
  const unsigned short* xr = xbf + (size_t)tok[gRow] * HID;

  const int am0 = waveM * 32 + l15;  // A rows in LDS tile for this lane
  const int am1 = am0 + 16;

  for (int k0 = 0; k0 < HID; k0 += 256) {
#pragma unroll
    for (int i = 0; i < 128; i += 8) {
      const uint4 q = *(const uint4*)(xr + k0 + sc + i);
      *(uint4*)&sxu[sr][(sc + i) >> 1] = q;
    }
    __syncthreads();
    if (k0 + 256 < HID) {  // prefetch next weight chunk (global_prefetch_b8)
      __builtin_prefetch(w1r0 + k0 + 256 + half * 16, 0, 1);
      __builtin_prefetch(w1r1 + k0 + 256 + half * 16, 0, 1);
      __builtin_prefetch(w3r0 + k0 + 256 + half * 16, 0, 1);
      __builtin_prefetch(w3r1 + k0 + 256 + half * 16, 0, 1);
    }
#pragma unroll
    for (int kk = 0; kk < 256; kk += 32) {
      FragU a0, a1;
      a0.q[0] = *(const uint4*)&sxu[am0][(kk + half * 16) >> 1];
      a0.q[1] = *(const uint4*)&sxu[am0][(kk + 8 + half * 16) >> 1];
      a1.q[0] = *(const uint4*)&sxu[am1][(kk + half * 16) >> 1];
      a1.q[1] = *(const uint4*)&sxu[am1][(kk + 8 + half * 16) >> 1];
      const int kb = k0 + kk + half * 16;
      FragU b;
      b = load_b_row(w1r0, kb);
      c1m0n0 = WMMA_BF16(a0.v, b.v, c1m0n0);
      c1m1n0 = WMMA_BF16(a1.v, b.v, c1m1n0);
      b = load_b_row(w1r1, kb);
      c1m0n1 = WMMA_BF16(a0.v, b.v, c1m0n1);
      c1m1n1 = WMMA_BF16(a1.v, b.v, c1m1n1);
      b = load_b_row(w3r0, kb);
      c3m0n0 = WMMA_BF16(a0.v, b.v, c3m0n0);
      c3m1n0 = WMMA_BF16(a1.v, b.v, c3m1n0);
      b = load_b_row(w3r1, kb);
      c3m0n1 = WMMA_BF16(a0.v, b.v, c3m0n1);
      c3m1n1 = WMMA_BF16(a1.v, b.v, c3m1n1);
    }
    __syncthreads();
  }

  // epilogue: silu(u1)*u3, store bf16 h (compacted rows)
#pragma unroll
  for (int j = 0; j < 8; ++j) {
    const int rbase = waveM * 32 + half * 8 + j;
    // M subtile 0
    int r = rbase;
    if (t0 + r < cnt) {
      __bf16* hp = h + (size_t)(goff + r) * FFN_;
      float u = c1m0n0[j];
      hp[n0] = (__bf16)((u / (1.f + __expf(-u))) * c3m0n0[j]);
      u = c1m0n1[j];
      hp[n1] = (__bf16)((u / (1.f + __expf(-u))) * c3m0n1[j]);
    }
    // M subtile 1
    r = rbase + 16;
    if (t0 + r < cnt) {
      __bf16* hp = h + (size_t)(goff + r) * FFN_;
      float u = c1m1n0[j];
      hp[n0] = (__bf16)((u / (1.f + __expf(-u))) * c3m1n0[j]);
      u = c1m1n1[j];
      hp[n1] = (__bf16)((u / (1.f + __expf(-u))) * c3m1n1[j]);
    }
  }
}

// ---------------------------------------------------------------------------
// K6: down projection.  out[tok] += gate * (h @ w2e^T)
// block tile: 128 tokens x 64 HID, same wave layout as K5.
// ---------------------------------------------------------------------------
__global__ __launch_bounds__(256) void moe_down(const unsigned short* __restrict__ h,
                                                const float* __restrict__ w2,
                                                const int* __restrict__ cnts,
                                                const int* __restrict__ offs,
                                                const int* __restrict__ tok,
                                                const float* __restrict__ gw,
                                                float* __restrict__ out) {
  const int e   = blockIdx.z;
  const int cnt = cnts[e];
  const int t0  = blockIdx.y * 128;
  if (t0 >= cnt) return;
  const int goff = offs[e] + t0;
  const int db   = blockIdx.x * 64;

  const int tid  = threadIdx.x;
  const int lane = tid & 31, wid = tid >> 5;
  const int l15 = lane & 15, half = lane >> 4;
  const int waveM = wid >> 1, waveN = wid & 1;

  __shared__ unsigned shu[128][128];  // 128 rows x 256 bf16 of h-chunk (64KB)

  const size_t wbase = (size_t)e * HID * FFN_;
  const int n0 = db + waveN * 32 + l15;
  const int n1 = n0 + 16;
  const float* w2r0 = w2 + wbase + (size_t)n0 * FFN_;
  const float* w2r1 = w2 + wbase + (size_t)n1 * FFN_;

  v8f cm0n0 = {}, cm1n0 = {}, cm0n1 = {}, cm1n1 = {};

  const int sr = tid >> 1;
  const int sc = (tid & 1) * 128;
  int gRow = goff + sr;
  if (t0 + sr >= cnt) gRow = goff;
  const unsigned short* hr = h + (size_t)gRow * FFN_;

  const int am0 = waveM * 32 + l15;
  const int am1 = am0 + 16;

  for (int k0 = 0; k0 < FFN_; k0 += 256) {
#pragma unroll
    for (int i = 0; i < 128; i += 8) {
      const uint4 q = *(const uint4*)(hr + k0 + sc + i);
      *(uint4*)&shu[sr][(sc + i) >> 1] = q;
    }
    __syncthreads();
    if (k0 + 256 < FFN_) {
      __builtin_prefetch(w2r0 + k0 + 256 + half * 16, 0, 1);
      __builtin_prefetch(w2r1 + k0 + 256 + half * 16, 0, 1);
    }
#pragma unroll
    for (int kk = 0; kk < 256; kk += 32) {
      FragU a0, a1;
      a0.q[0] = *(const uint4*)&shu[am0][(kk + half * 16) >> 1];
      a0.q[1] = *(const uint4*)&shu[am0][(kk + 8 + half * 16) >> 1];
      a1.q[0] = *(const uint4*)&shu[am1][(kk + half * 16) >> 1];
      a1.q[1] = *(const uint4*)&shu[am1][(kk + 8 + half * 16) >> 1];
      const int kb = k0 + kk + half * 16;
      FragU b;
      b = load_b_row(w2r0, kb);
      cm0n0 = WMMA_BF16(a0.v, b.v, cm0n0);
      cm1n0 = WMMA_BF16(a1.v, b.v, cm1n0);
      b = load_b_row(w2r1, kb);
      cm0n1 = WMMA_BF16(a0.v, b.v, cm0n1);
      cm1n1 = WMMA_BF16(a1.v, b.v, cm1n1);
    }
    __syncthreads();
  }

#pragma unroll
  for (int j = 0; j < 8; ++j) {
    const int rbase = waveM * 32 + half * 8 + j;
    int r = rbase;
    if (t0 + r < cnt) {
      const int g = goff + r;
      const int tk = tok[g];
      const float wsc = gw[g];
      unsafeAtomicAdd(out + (size_t)tk * HID + n0, cm0n0[j] * wsc);
      unsafeAtomicAdd(out + (size_t)tk * HID + n1, cm0n1[j] * wsc);
    }
    r = rbase + 16;
    if (t0 + r < cnt) {
      const int g = goff + r;
      const int tk = tok[g];
      const float wsc = gw[g];
      unsafeAtomicAdd(out + (size_t)tk * HID + n0, cm1n0[j] * wsc);
      unsafeAtomicAdd(out + (size_t)tk * HID + n1, cm1n1[j] * wsc);
    }
  }
}

// ---------------------------------------------------------------------------
extern "C" void kernel_launch(void* const* d_in, const int* in_sizes, int n_in,
                              void* d_out, int out_size, void* d_ws, size_t ws_size,
                              hipStream_t stream) {
  const float* x  = (const float*)d_in[0];  // [2,1024,2048]
  const float* wg = (const float*)d_in[1];  // [8,2048]
  const float* w1 = (const float*)d_in[2];  // [8,5632,2048]
  const float* w2 = (const float*)d_in[3];  // [8,2048,5632]
  const float* w3 = (const float*)d_in[4];  // [8,5632,2048]
  float* out = (float*)d_out;

  char* ws = (char*)d_ws;
  size_t off = 0;
  __bf16* hbuf = (__bf16*)(ws + off);                off += (size_t)MAXROWS * FFN_ * 2;  // 46.1 MB
  unsigned short* xbf = (unsigned short*)(ws + off); off += (size_t)NTOK * HID * 2;      // 8.4 MB
  int*   cnts  = (int*)(ws + off);   off += 256;
  int*   offsv = (int*)(ws + off);   off += 256;
  int*   tok   = (int*)(ws + off);   off += (size_t)MAXROWS * 4;
  float* gw    = (float*)(ws + off); off += (size_t)MAXROWS * 4;
  int*   sel   = (int*)(ws + off);   off += (size_t)MAXROWS * 4;
  float* selw  = (float*)(ws + off); off += (size_t)MAXROWS * 4;

  moe_router<<<NTOK / 8, 256, 0, stream>>>(x, wg, sel, selw);
  moe_build<<<1, 256, 0, stream>>>(sel, selw, cnts, offsv, tok, gw);
  moe_zero<<<(NTOK * HID) / 1024, 256, 0, stream>>>(out);
  moe_cvt_x<<<(NTOK * HID) / 2048, 256, 0, stream>>>(x, (unsigned*)xbf);

  dim3 gUp(FFN_ / 64, NTOK / 128, NEXP);
  moe_up<<<gUp, 256, 0, stream>>>(xbf, w1, w3, cnts, offsv, tok, hbuf);

  dim3 gDn(HID / 64, NTOK / 128, NEXP);
  moe_down<<<gDn, 256, 0, stream>>>((const unsigned short*)hbuf, w2, cnts, offsv, tok, gw, out);
}
